// GatedExpert_40484361732516
// MI455X (gfx1250) — matrix operand order repaired and checked by reading, not compiled, not run on hardware
//
#include <hip/hip_runtime.h>
#include <hip/hip_bf16.h>

// ---------------------------------------------------------------------------
// MoE gated-autoencoder forward on gfx1250 (MI455X), bf16 WMMA everywhere.
// GEMM: 128x128 block tile, 8 wave32s (4Mx2N), wave tile 32x64 (2x4 WMMA),
// K step 32, double-buffered LDS filled by GLOBAL_LOAD_ASYNC_TO_LDS_B128
// (ASYNCcnt-tracked, no VGPR staging), 1 barrier/step, k+2 prefetch.
// ---------------------------------------------------------------------------

typedef __attribute__((ext_vector_type(16))) __bf16 v16bf;
typedef __attribute__((ext_vector_type(8)))  float  v8f;
typedef int v4i_vs __attribute__((vector_size(16)));   // matches builtin proto

#define AS1 __attribute__((address_space(1)))
#define AS3 __attribute__((address_space(3)))

#if defined(__AMDGCN__) && __has_builtin(__builtin_amdgcn_global_load_async_to_lds_b128)
#define HAS_ASYNC_LDS 1
#else
#define HAS_ASYNC_LDS 0
#endif

#define NE 8
#define NB 4096
#define ND 3072
#define NH 1024
#define NL 512
#define NC 100
#define NCP 128          // padded classes (multiple of 128)

// ---------------- f32 -> bf16 convert ----------------
__global__ void cvt_bf16_kernel(const float* __restrict__ src,
                                __bf16* __restrict__ dst, long n) {
    long i = (long)blockIdx.x * blockDim.x + threadIdx.x;
    if (i < n) dst[i] = (__bf16)src[i];
}

// ---------------- transpose+convert: W[e][K][N] f32 -> Wt[e][Npad][K] bf16 ----
__global__ __launch_bounds__(256)
void transpose_bf16_kernel(const float* __restrict__ W, __bf16* __restrict__ Wt,
                           int K, int N, int Npad) {
    __shared__ float tile[32][33];
    const int e  = blockIdx.z;
    const float* We  = W  + (long)e * K * N;
    __bf16*      Wte = Wt + (long)e * Npad * K;
    const int n0 = blockIdx.x * 32, k0 = blockIdx.y * 32;
    const int tx = threadIdx.x & 31, ty = threadIdx.x >> 5;   // 32 x 8
#pragma unroll
    for (int i = 0; i < 32; i += 8) {
        int k = k0 + ty + i, n = n0 + tx;
        tile[ty + i][tx] = (n < N) ? We[(long)k * N + n] : 0.0f;
    }
    __syncthreads();
#pragma unroll
    for (int i = 0; i < 32; i += 8) {
        int n = n0 + ty + i, k = k0 + tx;
        Wte[(long)n * K + k] = (__bf16)tile[tx][ty + i];      // zero-padded rows
    }
}

// ---------------- tiled bf16 WMMA GEMM: Out = act(A[e] * Wt[e]^T + bias[e]) --
// A  : [NB x K]   bf16 row-major (per-expert stride strideAe, may be 0/shared)
// Wt : [Npad x K] bf16 row-major (weights pre-transposed)
#define LDSPITCH 56   // bf16 elems per LDS row (112B: 16B aligned, conflict-free)

template<bool RELU, bool OUT_BF16, bool OUT_F32>
__global__ __launch_bounds__(256, 1)
void gemm_bf16_wmma(const __bf16* __restrict__ A, long strideAe, int lda,
                    const __bf16* __restrict__ Wt, long strideWe,
                    const float*  __restrict__ bias, int Nreal,
                    __bf16* __restrict__ Ob, long strideObE, int ldob,
                    float*  __restrict__ Of, long strideOfE, int ldof,
                    int K) {
    __shared__ alignas(16) __bf16 Ash[2][128 * LDSPITCH];
    __shared__ alignas(16) __bf16 Bsh[2][128 * LDSPITCH];

    const int e    = blockIdx.z;
    const __bf16* Ae  = A  + (long)e * strideAe;
    const __bf16* Wte = Wt + (long)e * strideWe;
    const float*  be  = bias + (long)e * Nreal;

    const int tid  = threadIdx.x;
    const int lane = tid & 31;
    const int wave = tid >> 5;
    const int wm   = wave >> 1;          // 4 waves along M (32 rows each)
    const int wn   = wave & 1;           // 2 waves along N (64 cols each)
    const int row0 = blockIdx.y * 128;
    const int col0 = blockIdx.x * 128;
    const int half = lane >> 4;          // 0: lanes 0-15, 1: lanes 16-31
    const int mr   = lane & 15;
    const int cr   = tid >> 2;           // staging row 0..63
    const int ckc  = (tid & 3) * 8;      // staging K offset {0,8,16,24}

    v8f acc[2][4] = {};

    // Fragments per ISA 7.12.2 layouts; B-frag loaded one at a time to keep
    // register pressure low.
    union FragU { uint4 u[2]; v16bf v; };
    auto compute = [&](int buf) {
        v16bf afrag[2];
#pragma unroll
        for (int mi = 0; mi < 2; ++mi) {
            const __bf16* p = &Ash[buf][(wm * 32 + mi * 16 + mr) * LDSPITCH];
            FragU fu;
            fu.u[0] = *(const uint4*)&p[8 * half];        // K 0-7  (+8 if half)
            fu.u[1] = *(const uint4*)&p[16 + 8 * half];   // K 16-23(+8 if half)
            afrag[mi] = fu.v;
        }
#pragma unroll
        for (int ni = 0; ni < 4; ++ni) {
            const __bf16* p = &Bsh[buf][(wn * 64 + ni * 16 + mr) * LDSPITCH];
            FragU fu;
            fu.u[0] = *(const uint4*)&p[8 * half];        // K 0-7  (+8 if half)
            fu.u[1] = *(const uint4*)&p[16 + 8 * half];   // K 16-23(+8 if half)
            const v16bf bfrag = fu.v;
            acc[0][ni] = __builtin_amdgcn_wmma_f32_16x16x32_bf16(
                false, afrag[0], false, bfrag, (short)0, acc[0][ni], false, false);
            acc[1][ni] = __builtin_amdgcn_wmma_f32_16x16x32_bf16(
                false, afrag[1], false, bfrag, (short)0, acc[1][ni], false, false);
        }
    };

    const int nk = K >> 5;

#if HAS_ASYNC_LDS
    // ---- async global->LDS tile staging: no VGPR data, ASYNCcnt tracked ----
    auto gasync = [&](int kt, int buf) {
        const long kk = (long)kt * 32 + ckc;
        __builtin_amdgcn_global_load_async_to_lds_b128(
            (AS1 v4i_vs*)&Ae[(long)(row0 + cr) * lda + kk],
            (AS3 v4i_vs*)&Ash[buf][cr * LDSPITCH + ckc], 0, 0);
        __builtin_amdgcn_global_load_async_to_lds_b128(
            (AS1 v4i_vs*)&Ae[(long)(row0 + cr + 64) * lda + kk],
            (AS3 v4i_vs*)&Ash[buf][(cr + 64) * LDSPITCH + ckc], 0, 0);
        __builtin_amdgcn_global_load_async_to_lds_b128(
            (AS1 v4i_vs*)&Wte[(long)(col0 + cr) * K + kk],
            (AS3 v4i_vs*)&Bsh[buf][cr * LDSPITCH + ckc], 0, 0);
        __builtin_amdgcn_global_load_async_to_lds_b128(
            (AS1 v4i_vs*)&Wte[(long)(col0 + cr + 64) * K + kk],
            (AS3 v4i_vs*)&Bsh[buf][(cr + 64) * LDSPITCH + ckc], 0, 0);
    };
    auto wait_async = [&]() {
#if __has_builtin(__builtin_amdgcn_s_wait_asynccnt)
        __builtin_amdgcn_s_wait_asynccnt(0);
#else
        asm volatile("s_wait_asynccnt 0" ::: "memory");
#endif
    };

    gasync(0, 0);
    wait_async();
    __syncthreads();
    for (int kt = 0; kt < nk - 1; ++kt) {
        gasync(kt + 1, (kt & 1) ^ 1);            // fill next buffer while computing
        if (kt + 2 < nk) {                       // pull k+2 lines toward the WGP
            const long kk = (long)(kt + 2) * 32 + ckc;
            __builtin_prefetch(&Ae [(long)(row0 + cr) * lda + kk], 0, 0);
            __builtin_prefetch(&Wte[(long)(col0 + cr) * K   + kk], 0, 0);
        }
        compute(kt & 1);
        wait_async();                            // async writes visible...
        __syncthreads();                         // ...to all waves
    }
    compute((nk - 1) & 1);
#else
    // ---- fallback: register-staged double buffering ----
    uint4 ra[2], rb[2];
    auto gload = [&](int kt) {
        const long kk = (long)kt * 32 + ckc;
        ra[0] = *(const uint4*)&Ae [(long)(row0 + cr)      * lda + kk];
        ra[1] = *(const uint4*)&Ae [(long)(row0 + cr + 64) * lda + kk];
        rb[0] = *(const uint4*)&Wte[(long)(col0 + cr)      * K   + kk];
        rb[1] = *(const uint4*)&Wte[(long)(col0 + cr + 64) * K   + kk];
    };
    auto lstore = [&](int buf) {
        *(uint4*)&Ash[buf][cr * LDSPITCH + ckc]        = ra[0];
        *(uint4*)&Ash[buf][(cr + 64) * LDSPITCH + ckc] = ra[1];
        *(uint4*)&Bsh[buf][cr * LDSPITCH + ckc]        = rb[0];
        *(uint4*)&Bsh[buf][(cr + 64) * LDSPITCH + ckc] = rb[1];
    };
    gload(0);
    lstore(0);
    __syncthreads();
    for (int kt = 0; kt < nk - 1; ++kt) {
        gload(kt + 1);
        compute(kt & 1);
        lstore((kt & 1) ^ 1);
        __syncthreads();
    }
    compute((nk - 1) & 1);
#endif

    // ---- epilogue: bias (+relu), bf16 and/or f32 stores ----
#pragma unroll
    for (int mi = 0; mi < 2; ++mi) {
#pragma unroll
        for (int ni = 0; ni < 4; ++ni) {
            const int n  = col0 + wn * 64 + ni * 16 + mr;
            const float bv = (n < Nreal) ? be[n] : 0.0f;
#pragma unroll
            for (int r = 0; r < 8; ++r) {
                const int m = row0 + wm * 32 + mi * 16 + r + 8 * half;
                float v = acc[mi][ni][r] + bv;
                if (RELU) v = fmaxf(v, 0.0f);
                if (OUT_BF16)
                    Ob[(long)e * strideObE + (long)m * ldob + n] = (__bf16)v;
                if (OUT_F32)
                    Of[(long)e * strideOfE + (long)m * ldof + n] = v;
            }
        }
    }
}

// ---------------- per-(e,b) L1 reconstruction error ----------------
__global__ __launch_bounds__(256)
void l1err_kernel(const float* __restrict__ recon, const float* __restrict__ x,
                  float* __restrict__ err) {
    const long eb = blockIdx.x;
    const long b  = eb & (NB - 1);
    const float* r  = recon + eb * ND;
    const float* xr = x + b * ND;
    float s = 0.0f;
    for (int i = threadIdx.x; i < ND; i += 256) s += fabsf(r[i] - xr[i]);
    __shared__ float red[256];
    red[threadIdx.x] = s;
    __syncthreads();
    for (int o = 128; o > 0; o >>= 1) {
        if (threadIdx.x < o) red[threadIdx.x] += red[threadIdx.x + o];
        __syncthreads();
    }
    if (threadIdx.x == 0) err[eb] = red[0] * (1.0f / (float)ND);
}

// ---------------- softmax over experts, argmin, mask ----------------
__global__ void select_kernel(const float* __restrict__ err,
                              float* __restrict__ indices_f,
                              float* __restrict__ min_err,
                              float* __restrict__ relevance,
                              float* __restrict__ mask,
                              int* __restrict__ idx_i) {
    int b = blockIdx.x * blockDim.x + threadIdx.x;
    if (b >= NB) return;
    float ev[NE];
#pragma unroll
    for (int e = 0; e < NE; ++e) ev[e] = err[e * NB + b];
    float mn = ev[0]; int am = 0;
#pragma unroll
    for (int e = 1; e < NE; ++e) if (ev[e] < mn) { mn = ev[e]; am = e; }
    float smax = -1e30f;
#pragma unroll
    for (int e = 0; e < NE; ++e) smax = fmaxf(smax, -ev[e] * 0.5f);
    float den = 0.0f;
#pragma unroll
    for (int e = 0; e < NE; ++e) den += __expf(-ev[e] * 0.5f - smax);
    const float rden = 1.0f / den;
#pragma unroll
    for (int e = 0; e < NE; ++e) {
        relevance[e * NB + b] = __expf(-ev[e] * 0.5f - smax) * rden;
        mask[e * NB + b] = (e == am) ? 1.0f : 0.0f;
    }
    indices_f[b] = (float)am;
    min_err[b]   = mn;
    idx_i[b]     = am;
}

// ---------------- gather winning expert logits ----------------
__global__ void gather_logits_kernel(const float* __restrict__ exp_out,
                                     const int* __restrict__ idx,
                                     float* __restrict__ logits) {
    int i = blockIdx.x * blockDim.x + threadIdx.x;
    if (i >= NB * NC) return;
    int b = i / NC, c = i - b * NC;
    logits[i] = exp_out[((long)idx[b] * NB + b) * NCP + c];
}

// ---------------------------------------------------------------------------
extern "C" void kernel_launch(void* const* d_in, const int* in_sizes, int n_in,
                              void* d_out, int out_size, void* d_ws, size_t ws_size,
                              hipStream_t stream) {
    (void)in_sizes; (void)n_in; (void)out_size; (void)ws_size;
    const float* x      = (const float*)d_in[0];
    const float* enc_w0 = (const float*)d_in[1];  const float* enc_b0 = (const float*)d_in[2];
    const float* enc_w1 = (const float*)d_in[3];  const float* enc_b1 = (const float*)d_in[4];
    const float* enc_w2 = (const float*)d_in[5];  const float* enc_b2 = (const float*)d_in[6];
    const float* dec_w0 = (const float*)d_in[7];  const float* dec_b0 = (const float*)d_in[8];
    const float* dec_w1 = (const float*)d_in[9];  const float* dec_b1 = (const float*)d_in[10];
    const float* dec_w2 = (const float*)d_in[11]; const float* dec_b2 = (const float*)d_in[12];
    const float* exp_w0 = (const float*)d_in[13]; const float* exp_b0 = (const float*)d_in[14];
    const float* exp_w1 = (const float*)d_in[15]; const float* exp_b1 = (const float*)d_in[16];
    const float* exp_w2 = (const float*)d_in[17]; const float* exp_b2 = (const float*)d_in[18];

    // ---- workspace layout (256B aligned) ----
    char* ws = (char*)d_ws;
    size_t off = 0;
    auto alloc = [&](size_t bytes) -> size_t {
        size_t o = off; off += (bytes + 255) & ~(size_t)255; return o;
    };
    const size_t o_xb   = alloc((size_t)NB * ND * 2);
    const size_t o_tew0 = alloc((size_t)NE * NH * ND * 2);
    const size_t o_tew1 = alloc((size_t)NE * NH * NH * 2);
    const size_t o_tew2 = alloc((size_t)NE * NL * NH * 2);
    const size_t o_tdw0 = alloc((size_t)NE * NH * NL * 2);
    const size_t o_tdw1 = alloc((size_t)NE * NH * NH * 2);
    const size_t o_tdw2 = alloc((size_t)NE * ND * NH * 2);
    const size_t o_txw0 = alloc((size_t)NE * NH * NL * 2);
    const size_t o_txw1 = alloc((size_t)NE * NH * NH * 2);
    const size_t o_txw2 = alloc((size_t)NE * NCP * NH * 2);
    const size_t o_actA = alloc((size_t)NE * NB * NH * 2);
    const size_t o_actB = alloc((size_t)NE * NB * NH * 2);
    const size_t o_lat  = alloc((size_t)NE * NB * NL * 2);
    const size_t o_expo = alloc((size_t)NE * NB * NCP * 4);
    const size_t o_err  = alloc((size_t)NE * NB * 4);
    const size_t o_idx  = alloc((size_t)NB * 4);

    __bf16* xb    = (__bf16*)(ws + o_xb);
    __bf16* tew0  = (__bf16*)(ws + o_tew0);
    __bf16* tew1  = (__bf16*)(ws + o_tew1);
    __bf16* tew2  = (__bf16*)(ws + o_tew2);
    __bf16* tdw0  = (__bf16*)(ws + o_tdw0);
    __bf16* tdw1  = (__bf16*)(ws + o_tdw1);
    __bf16* tdw2  = (__bf16*)(ws + o_tdw2);
    __bf16* txw0  = (__bf16*)(ws + o_txw0);
    __bf16* txw1  = (__bf16*)(ws + o_txw1);
    __bf16* txw2  = (__bf16*)(ws + o_txw2);
    __bf16* actA  = (__bf16*)(ws + o_actA);
    __bf16* actB  = (__bf16*)(ws + o_actB);
    __bf16* lat   = (__bf16*)(ws + o_lat);
    float*  expo  = (float*)(ws + o_expo);
    float*  err   = (float*)(ws + o_err);
    int*    idx_i = (int*)(ws + o_idx);

    // ---- output slices (d_out is float, tuple concatenated flat) ----
    float* out      = (float*)d_out;
    float* o_logits = out;
    float* o_recon  = o_logits + (size_t)NB * NC;
    float* o_ind    = o_recon + (size_t)NE * NB * ND;
    float* o_minerr = o_ind + NB;
    float* o_rel    = o_minerr + NB;
    float* o_mask   = o_rel + (size_t)NE * NB;

    // 1) x -> bf16
    {
        long n = (long)NB * ND;
        cvt_bf16_kernel<<<dim3((unsigned)((n + 255) / 256)), 256, 0, stream>>>(x, xb, n);
    }
    // 2) transpose+convert weights: W[e][K][N] -> Wt[e][Npad][K]
    auto launch_tr = [&](const float* W, __bf16* Wt, int K, int N, int Npad) {
        transpose_bf16_kernel<<<dim3(Npad / 32, K / 32, NE), 256, 0, stream>>>(W, Wt, K, N, Npad);
    };
    launch_tr(enc_w0, tew0, ND, NH, NH);
    launch_tr(enc_w1, tew1, NH, NH, NH);
    launch_tr(enc_w2, tew2, NH, NL, NL);
    launch_tr(dec_w0, tdw0, NL, NH, NH);
    launch_tr(dec_w1, tdw1, NH, NH, NH);
    launch_tr(dec_w2, tdw2, NH, ND, ND);
    launch_tr(exp_w0, txw0, NL, NH, NH);
    launch_tr(exp_w1, txw1, NH, NH, NH);
    launch_tr(exp_w2, txw2, NH, NC, NCP);

    const dim3 blk(256);
    // GEMM grid: grid.x = Npad/128, grid.y = NB/128, grid.z = NE
    auto grid_for = [](int Npad) { return dim3(Npad / 128, NB / 128, NE); };

    // 3) encoder: h0 = relu(xb @ W0^T+b), h1 = relu(...), lat = h1@W2^T+b
    gemm_bf16_wmma<true, true, false><<<grid_for(NH), blk, 0, stream>>>(
        xb, 0L, ND, tew0, (long)NH * ND, enc_b0, NH,
        actA, (long)NB * NH, NH, nullptr, 0L, 0, ND);
    gemm_bf16_wmma<true, true, false><<<grid_for(NH), blk, 0, stream>>>(
        actA, (long)NB * NH, NH, tew1, (long)NH * NH, enc_b1, NH,
        actB, (long)NB * NH, NH, nullptr, 0L, 0, NH);
    gemm_bf16_wmma<false, true, false><<<grid_for(NL), blk, 0, stream>>>(
        actB, (long)NB * NH, NH, tew2, (long)NL * NH, enc_b2, NL,
        lat, (long)NB * NL, NL, nullptr, 0L, 0, NH);
    // 4) decoder: recon (f32, straight into d_out reconstructions slice)
    gemm_bf16_wmma<true, true, false><<<grid_for(NH), blk, 0, stream>>>(
        lat, (long)NB * NL, NL, tdw0, (long)NH * NL, dec_b0, NH,
        actA, (long)NB * NH, NH, nullptr, 0L, 0, NL);
    gemm_bf16_wmma<true, true, false><<<grid_for(NH), blk, 0, stream>>>(
        actA, (long)NB * NH, NH, tdw1, (long)NH * NH, dec_b1, NH,
        actB, (long)NB * NH, NH, nullptr, 0L, 0, NH);
    gemm_bf16_wmma<false, false, true><<<grid_for(ND), blk, 0, stream>>>(
        actB, (long)NB * NH, NH, tdw2, (long)ND * NH, dec_b2, ND,
        nullptr, 0L, 0, o_recon, (long)NB * ND, ND, NH);
    // 5) L1 error + gating
    l1err_kernel<<<dim3(NE * NB), 256, 0, stream>>>(o_recon, x, err);
    select_kernel<<<dim3(NB / 256), 256, 0, stream>>>(err, o_ind, o_minerr,
                                                      o_rel, o_mask, idx_i);
    // 6) experts (dense), then gather winner
    gemm_bf16_wmma<true, true, false><<<grid_for(NH), blk, 0, stream>>>(
        lat, (long)NB * NL, NL, txw0, (long)NH * NL, exp_b0, NH,
        actA, (long)NB * NH, NH, nullptr, 0L, 0, NL);
    gemm_bf16_wmma<true, true, false><<<grid_for(NH), blk, 0, stream>>>(
        actA, (long)NB * NH, NH, txw1, (long)NH * NH, exp_b1, NH,
        actB, (long)NB * NH, NH, nullptr, 0L, 0, NH);
    gemm_bf16_wmma<false, false, true><<<grid_for(NCP), blk, 0, stream>>>(
        actB, (long)NB * NH, NH, txw2, (long)NCP * NH, exp_b2, NC,
        nullptr, 0L, 0, expo, (long)NB * NCP, NCP, NH);
    gather_logits_kernel<<<dim3((NB * NC + 255) / 256), 256, 0, stream>>>(
        expo, idx_i, o_logits);
}
